// SGN_60052232732880
// MI455X (gfx1250) — matrix operand order repaired and checked by reading, not compiled, not run on hardware
//
#include <hip/hip_runtime.h>

#define EPS 1e-5f

typedef __bf16 bf16;
typedef bf16 v16bf __attribute__((ext_vector_type(16)));
typedef bf16 v8bf  __attribute__((ext_vector_type(8)));
typedef float v8f  __attribute__((ext_vector_type(8)));

// ---------------------------------------------------------------------------
// WMMA bf16 GEMM for conv1x1:  Y[n, m] = post-ops( sum_k W[m,k]*X[n,k] )
// epilogue: acc (+pre) -> *scale + shift -> relu? -> +post -> +tem[n%20][m]
// Each wave: one 16-row M tile x eight 16-col N tiles (A reused 8x per k-step).
// grid: (N/1024, M/16), block 256 (8 waves)
// ---------------------------------------------------------------------------
__global__ __launch_bounds__(256) void k_wmma_conv(
    const bf16* __restrict__ X, int ldx,
    const bf16* __restrict__ W,          // pre-converted bf16, row-major [M][K]
    const float* __restrict__ scale,
    const float* __restrict__ shift,
    bf16* Y, int ldy,
    const bf16* Pre, const bf16* Post,
    const float* __restrict__ Tem,
    int K, int M, int relu)
{
  const int lane = threadIdx.x & 31;
  const int wv   = threadIdx.x >> 5;
  const int n0 = (blockIdx.x * 8 + wv) * 128;
  const int m0 = blockIdx.y * 16;
  const int ml = lane & 15, hi = lane >> 4;

  const bf16* wr = W + (size_t)(m0 + ml) * K;
  const bf16* xr = X + (size_t)(n0 + ml) * ldx;
  const size_t nstride = (size_t)16 * ldx;

  v8f acc[8] = {};
  for (int k0 = 0; k0 < K; k0 += 32) {
    // A tile: 16 bf16 of W row, K-contiguous (k = 16*hi + s), matching B
    v8bf a0 = *(const v8bf*)(wr + k0 + 16*hi);
    v8bf a1 = *(const v8bf*)(wr + k0 + 16*hi + 8);
    v16bf a;
    #pragma unroll
    for (int i = 0; i < 8; ++i) { a[i] = a0[i]; a[8+i] = a1[i]; }
    #pragma unroll
    for (int j = 0; j < 8; ++j) {
      const bf16* xp = xr + j*nstride + k0 + 16*hi;   // B: K-contiguous bf16
      v8bf bl = *(const v8bf*)xp;
      v8bf bh = *(const v8bf*)(xp + 8);
      v16bf b;
      #pragma unroll
      for (int i = 0; i < 8; ++i) { b[i] = bl[i]; b[8+i] = bh[i]; }
      acc[j] = __builtin_amdgcn_wmma_f32_16x16x32_bf16(false, a, false, b,
                                                       (short)0, acc[j], false, false);
    }
  }

  const int mb = m0 + 8*hi;
  #pragma unroll
  for (int j = 0; j < 8; ++j) {
    const int n = n0 + j*16 + ml;
    const size_t orow = (size_t)n * ldy + mb;
    v8bf pre = {}, post = {};
    if (Pre)  pre  = *(const v8bf*)(Pre + orow);
    if (Post) post = *(const v8bf*)(Post + orow);
    v8bf o;
    #pragma unroll
    for (int r = 0; r < 8; ++r) {
      float y = acc[j][r];
      if (Pre) y += (float)pre[r];
      int m = mb + r;
      float s = scale ? scale[m] : 1.f;
      float t = shift ? shift[m] : 0.f;
      y = y * s + t;
      if (relu) y = fmaxf(y, 0.f);
      if (Post) y += (float)post[r];
      if (Tem)  y += Tem[(n % 20) * M + m];
      o[r] = (bf16)y;
    }
    *(v8bf*)(Y + orow) = o;
  }
}

// fp32 -> bf16 weight conversion (once per layer)
__global__ void k_w2bf(const float* __restrict__ W, bf16* __restrict__ O, int total)
{
  int tid = blockIdx.x*blockDim.x + threadIdx.x;
  if (tid < total) O[tid] = (bf16)W[tid];
}

// ---------------------------------------------------------------------------
// Input prep: gather inp (B,3,V,T) + temporal diff, apply the two norm_data BNs
// ---------------------------------------------------------------------------
__global__ void k_prep(const float* __restrict__ x,
    const float* gj,const float* bj,const float* mj,const float* vj,
    const float* gd,const float* bd,const float* md,const float* vd,
    float* P3, float* D3, int total)
{
  int tid = blockIdx.x*blockDim.x + threadIdx.x;
  if (tid >= total) return;
  int c = tid % 3; int col = tid / 3;
  int b = col / 500; int p = col % 500; int v = p / 20; int t = p % 20;
  auto fetch = [&](int tq)->float {
    int j = tq*75 + v*3 + c;                 // reinterpret (3,20,25) as (20,25,3)
    int i = j / 500; int r = j % 500; int tt = r/25; int vv = r%25;
    return x[(((size_t)(b*3+i)*20+tt)*25+vv)*2];  // [...,0]
  };
  float val = fetch(t);
  float dif = (t>0) ? val - fetch(t-1) : 0.f;
  int ch = c*25 + v;                         // BN1d channel = c*V + v
  P3[(size_t)col*3+c] = (val - mj[ch]) * (gj[ch]*rsqrtf(vj[ch]+EPS)) + bj[ch];
  D3[(size_t)col*3+c] = (dif - md[ch]) * (gd[ch]*rsqrtf(vd[ch]+EPS)) + bd[ch];
}

// plain conv1x1 for K=3 (first embed layer), relu
__global__ void k_conv3(const float* __restrict__ X3, const float* __restrict__ W,
                        const float* __restrict__ Bs, bf16* __restrict__ E, int total)
{
  int tid = blockIdx.x*blockDim.x + threadIdx.x;
  if (tid >= total) return;
  int o = tid & 63; int col = tid >> 6;
  const float* xr = X3 + (size_t)col*3;
  float acc = Bs[o] + W[o*3]*xr[0] + W[o*3+1]*xr[1] + W[o*3+2]*xr[2];
  E[(size_t)col*64+o] = (bf16)fmaxf(acc, 0.f);
}

// tem/spa embeddings of identity inputs -> small lookup tables
__global__ void k_tables(const float* sw1,const float* sb1,const float* sw2,const float* sb2,
                         const float* tw1,const float* tb1,const float* tw2,const float* tb2,
                         float* spa_tab, float* tem_tab)
{
  int tid = blockIdx.x*blockDim.x + threadIdx.x;
  if (tid < 25*64) {
    int v = tid >> 6, o = tid & 63;
    float acc = sb2[o];
    for (int c = 0; c < 64; ++c) acc += sw2[o*64+c]*fmaxf(sw1[c*25+v]+sb1[c],0.f);
    spa_tab[v*64+o] = fmaxf(acc, 0.f);
  } else if (tid < 25*64 + 20*256) {
    int k = tid - 1600; int t = k >> 8, o = k & 255;
    float acc = tb2[o];
    for (int c = 0; c < 64; ++c) acc += tw2[o*64+c]*fmaxf(tw1[c*20+t]+tb1[c],0.f);
    tem_tab[t*256+o] = fmaxf(acc, 0.f);
  }
}

__global__ void k_fill_spa(const float* __restrict__ spa_tab, bf16* __restrict__ H, int total)
{
  int tid = blockIdx.x*blockDim.x + threadIdx.x;
  if (tid >= total) return;
  int o = tid & 63; int col = tid >> 6; int v = (col % 500) / 20;
  H[(size_t)col*128 + 64 + o] = (bf16)spa_tab[v*64+o];
}

// fold bias+BN into per-channel scale/shift
__global__ void k_bnfold(const float* g,const float* b,const float* m,const float* v,
                         const float* bias, float* scale, float* shift, int C)
{
  int c = blockIdx.x*blockDim.x + threadIdx.x;
  if (c >= C) return;
  float s = g[c]*rsqrtf(v[c]+EPS);
  scale[c] = s;
  shift[c] = b[c] - (m[c] - (bias ? bias[c] : 0.f))*s;
}

// per-(b,t) 25x25 attention matrix + softmax (+ accumulate previous G)
__global__ __launch_bounds__(256) void k_gmat(const bf16* __restrict__ A1,
    const bf16* __restrict__ A2, int ld, int C, float* G, int accum)
{
  __shared__ bf16 sA[6400]; __shared__ bf16 sB[6400]; __shared__ float sG[625];
  int bt = blockIdx.x; int b = bt/20, t = bt%20;
  int tid = threadIdx.x;
  for (int i = tid; i < 25*C; i += 256) {
    int v = i / C, c = i % C;
    size_t src = (size_t)(b*500 + v*20 + t)*ld + c;
    sA[i] = A1[src]; sB[i] = A2[src];
  }
  __syncthreads();
  for (int p = tid; p < 625; p += 256) {
    int v1 = p/25, v2 = p%25;
    float acc = 0.f;
    const bf16* ra = sA + v1*C; const bf16* rb = sB + v2*C;
    for (int c = 0; c < C; ++c) acc += (float)ra[c]*(float)rb[c];
    sG[p] = acc;
  }
  __syncthreads();
  if (tid < 25) {
    const float* row = sG + tid*25;
    float mx = -1e30f;
    for (int j = 0; j < 25; ++j) mx = fmaxf(mx, row[j]);
    float sum = 0.f;
    for (int j = 0; j < 25; ++j) sum += __expf(row[j]-mx);
    float inv = 1.f/sum;
    size_t gb = (size_t)bt*625 + tid*25;
    for (int j = 0; j < 25; ++j) {
      float val = __expf(row[j]-mx)*inv;
      if (accum) val += G[gb+j];
      G[gb+j] = val;
    }
  }
}

// message passing over joints: Out[n,c] = sum_v2 G[b,t,v,v2]*Hin[(b,v2,t),c]
__global__ void k_gx(const float* __restrict__ G, const bf16* __restrict__ Hin, int ldh,
                     int C, bf16* __restrict__ Out, int total)
{
  int tid = blockIdx.x*blockDim.x + threadIdx.x;
  if (tid >= total) return;
  int c = tid % C; int n = tid / C;
  int b = n/500, p = n%500, v = p/20, t = p%20;
  const float* gr = G + (size_t)(b*20+t)*625 + v*25;
  float acc = 0.f;
  for (int v2 = 0; v2 < 25; ++v2)
    acc += gr[v2] * (float)Hin[(size_t)(b*500 + v2*20 + t)*ldh + c];
  Out[(size_t)n*C + c] = (bf16)acc;
}

// SE gate: se = mean_t(Y); se1 = sigmoid(conv1d_V(se, k=25, pad=12) + b)
__global__ __launch_bounds__(256) void k_se_gate(const bf16* __restrict__ Y, int ld, int C,
    const float* __restrict__ saw, const float* __restrict__ sab, float* __restrict__ sgate)
{
  __shared__ float se[6400];
  int b = blockIdx.x; int tid = threadIdx.x;
  for (int i = tid; i < C*25; i += 256) {
    int c = i/25, v = i%25;
    float s = 0.f;
    for (int t = 0; t < 20; ++t) s += (float)Y[(size_t)(b*500+v*20+t)*ld + c];
    se[c*25+v] = s*(1.f/20.f);
  }
  __syncthreads();
  if (tid < 25) {
    float acc = 0.f;
    for (int c = 0; c < C; ++c) {
      const float* wr = saw + c*25; const float* sr = se + c*25;
      for (int k = 0; k < 25; ++k) {
        int vv = tid + k - 12;
        if (vv >= 0 && vv < 25) acc += wr[k]*sr[vv];
      }
    }
    acc += sab[0];
    sgate[b*25+tid] = 1.f/(1.f+__expf(-acc));
  }
}

// att output: Out = BN( F + Y*(1+sgate[b,v]) )
__global__ void k_att_out(const bf16* Fb, const bf16* Yv, const float* __restrict__ sgate,
    const float* g,const float* bb,const float* m,const float* vv,
    int C, bf16* Out, int total)
{
  int tid = blockIdx.x*blockDim.x + threadIdx.x;
  if (tid >= total) return;
  int c = tid % C; int n = tid / C;
  int b = n/500; int v = (n%500)/20;
  float x = (float)Fb[(size_t)n*C+c] + (float)Yv[(size_t)n*C+c]*(1.f + sgate[b*25+v]);
  float s = g[c]*rsqrtf(vv[c]+EPS);
  Out[(size_t)n*C+c] = (bf16)((x - m[c])*s + bb[c]);
}

// max over joints: Lt[(b,t),c] = max_v Ht[(b,v,t),c]
__global__ void k_pool(const bf16* __restrict__ Ht, bf16* __restrict__ Lt, int total)
{
  int tid = blockIdx.x*blockDim.x + threadIdx.x;
  if (tid >= total) return;
  int c = tid & 255; int bt = tid >> 8; int b = bt/20, t = bt%20;
  float mx = -1e30f;
  for (int v = 0; v < 25; ++v)
    mx = fmaxf(mx, (float)Ht[(size_t)(b*500+v*20+t)*256 + c]);
  Lt[(size_t)bt*256 + c] = (bf16)mx;
}

// temporal conv k=3 pad=1 + BN + relu
__global__ void k_cnn1(const bf16* __restrict__ Lt, const float* __restrict__ w,
    const float* __restrict__ cb,
    const float* g,const float* bb,const float* m,const float* vv,
    bf16* __restrict__ Z1, int total)
{
  int tid = blockIdx.x*blockDim.x + threadIdx.x;
  if (tid >= total) return;
  int o = tid & 255; int bt = tid >> 8; int b = bt/20, t = bt%20;
  float acc = cb[o];
  for (int dt = -1; dt <= 1; ++dt) {
    int tt = t + dt;
    if (tt < 0 || tt >= 20) continue;
    const bf16* xr = Lt + (size_t)(b*20+tt)*256;
    const float* wr = w + (size_t)o*768 + (dt+1);
    for (int i = 0; i < 256; ++i) acc += wr[i*3]*(float)xr[i];
  }
  float s = g[o]*rsqrtf(vv[o]+EPS);
  Z1[(size_t)bt*256+o] = (bf16)fmaxf((acc - m[o])*s + bb[o], 0.f);
}

// global max over T + FC 512->60
__global__ __launch_bounds__(512) void k_final(const bf16* __restrict__ L,
    const float* __restrict__ fw, const float* __restrict__ fb, float* __restrict__ out)
{
  __shared__ float sm[512];
  int b = blockIdx.x; int c = threadIdx.x;
  float mx = -1e30f;
  for (int t = 0; t < 20; ++t) mx = fmaxf(mx, (float)L[(size_t)(b*20+t)*512 + c]);
  sm[c] = mx;
  __syncthreads();
  if (c < 60) {
    float acc = fb[c];
    for (int k = 0; k < 512; ++k) acc += fw[c*512+k]*sm[k];
    out[b*60+c] = acc;
  }
}

// ---------------------------------------------------------------------------
extern "C" void kernel_launch(void* const* d_in, const int* in_sizes, int n_in,
                              void* d_out, int out_size, void* d_ws, size_t ws_size,
                              hipStream_t stream) {
  (void)n_in; (void)out_size; (void)ws_size;
  const int NB = 128000, N2 = 5120;
  auto F = [&](int i){ return (const float*)d_in[i]; };

  char* ws = (char*)d_ws;
  size_t off = 0;
  auto alloc = [&](size_t bytes)->char* {
    char* p = ws + off; off = (off + bytes + 255) & ~(size_t)255; return p;
  };
  float* fold    = (float*)alloc(13*4096);
  bf16*  wpool   = (bf16*)alloc((size_t)2*1024*1024*2); // bf16 weight pool
  float* P3      = (float*)alloc((size_t)NB*3*4);
  float* D3      = (float*)alloc((size_t)NB*3*4);
  float* spa_tab = (float*)alloc(25*64*4);
  float* tem_tab = (float*)alloc(20*256*4);
  float* sgate   = (float*)alloc(256*25*4);
  float* G       = (float*)alloc((size_t)5120*625*4);
  bf16* E64 = (bf16*)alloc((size_t)NB*64*2);
  bf16* Hb0 = (bf16*)alloc((size_t)NB*128*2);   // h1 (concat), preserved
  bf16* Hb1 = (bf16*)alloc((size_t)NB*256*2);
  bf16* Hb2 = (bf16*)alloc((size_t)NB*256*2);
  bf16* U1  = (bf16*)alloc((size_t)NB*256*2);   // A1 / Xg / F
  bf16* U2  = (bf16*)alloc((size_t)NB*256*2);   // A2 / T1
  bf16* Yb  = (bf16*)alloc((size_t)NB*256*2);
  bf16* Lt  = (bf16*)alloc((size_t)N2*256*2);
  bf16* Z1  = (bf16*)alloc((size_t)N2*256*2);
  bf16* Z2  = (bf16*)alloc((size_t)N2*512*2);
  bf16* Lf  = (bf16*)alloc((size_t)N2*512*2);

  // convert conv weights to bf16 once
  size_t woff = 0;
  auto wcvt = [&](int idx)->const bf16* {
    int elems = in_sizes[idx];
    bf16* p = wpool + woff;
    woff += ((size_t)elems + 127) & ~(size_t)127;
    k_w2bf<<<(elems+255)/256, 256, 0, stream>>>(F(idx), p, elems);
    return p;
  };
  const bf16 *Wj2 = wcvt(3),  *Wd2 = wcvt(11);
  const bf16 *Wg1a = wcvt(25), *Wg1b = wcvt(27);
  const bf16 *Wc1  = wcvt(37), *Wc1r = wcvt(38), *Wf1 = wcvt(44);
  const bf16 *Wg2a = wcvt(52), *Wg2b = wcvt(54);
  const bf16 *Wc2  = wcvt(64), *Wc2r = wcvt(65), *Wf2 = wcvt(71);
  const bf16 *Wg3a = wcvt(79), *Wg3b = wcvt(81);
  const bf16 *Wc3  = wcvt(91), *Wc3r = wcvt(92), *Wf3 = wcvt(98);
  const bf16 *Wres = wcvt(106), *Wta = wcvt(112);
  const bf16 *Wcn2 = wcvt(123), *Wlr = wcvt(129);

  auto conv = [&](const bf16* X, int ldx, const bf16* W, const float* sc, const float* sh,
                  bf16* Y, int ldy, const bf16* pre, const bf16* post, const float* tem,
                  int N, int K, int M, int relu) {
    dim3 g(N/1024, M/16);
    k_wmma_conv<<<g, 256, 0, stream>>>(X, ldx, W, sc, sh, Y, ldy, pre, post, tem, K, M, relu);
  };
  auto dofold = [&](int slot, int biasIdx, int gIdx) {
    float* sc = fold + slot*1024; float* sh = sc + 512;
    const float* bias = (biasIdx >= 0) ? F(biasIdx) : nullptr;
    k_bnfold<<<2, 256, 0, stream>>>(F(gIdx), F(gIdx+1), F(gIdx+2), F(gIdx+3), bias, sc, sh, 512);
  };
  auto fS = [&](int slot){ return fold + slot*1024; };
  auto fT = [&](int slot){ return fold + slot*1024 + 512; };

  // folded bias+BN params (only first C entries used per layer)
  dofold(0, 26, 29);   dofold(1, 28, 33);            // g1 bn1/bn2
  dofold(2, 39, 40);                                 // gcn1 bn
  dofold(3, 53, 56);   dofold(4, 55, 60);            // g2
  dofold(5, 66, 67);                                 // gcn2 bn
  dofold(6, 80, 83);   dofold(7, 82, 87);            // g3
  dofold(8, 93, 94);                                 // gcn3 bn
  dofold(9, 107, 108);                               // bn_res (+res_b)
  dofold(10, -1, 113);                               // tem_att_bn (no bias)
  dofold(11, 124, 125);                              // local bn2 (+cnn2_b)
  dofold(12, 130, 131);                              // local bn_re (+res_b)

  // ---- input prep + embeddings ----
  k_prep<<<(NB*3+255)/256, 256, 0, stream>>>((const float*)d_in[0],
      F(5),F(6),F(7),F(8), F(13),F(14),F(15),F(16), P3, D3, NB*3);
  k_tables<<<27, 256, 0, stream>>>(F(21),F(22),F(23),F(24), F(17),F(18),F(19),F(20),
                                   spa_tab, tem_tab);
  k_conv3<<<NB*64/256, 256, 0, stream>>>(P3, F(1), F(2), E64, NB*64);
  conv(E64, 64, Wj2, nullptr, F(4), Hb0, 128, nullptr, nullptr, nullptr, NB, 64, 64, 1);
  k_conv3<<<NB*64/256, 256, 0, stream>>>(D3, F(9), F(10), E64, NB*64);
  conv(E64, 64, Wd2, nullptr, F(12), Hb0, 128, nullptr, Hb0, nullptr, NB, 64, 64, 1);
  k_fill_spa<<<NB*64/256, 256, 0, stream>>>(spa_tab, Hb0, NB*64);

  // ---- gcn stage 1 (128 -> 128) ----
  conv(Hb0, 128, Wg1a, fS(0), fT(0), U1, 128, nullptr, nullptr, nullptr, NB, 128, 128, 0);
  conv(Hb0, 128, Wg1b, fS(1), fT(1), U2, 128, nullptr, nullptr, nullptr, NB, 128, 128, 0);
  k_gmat<<<5120, 256, 0, stream>>>(U1, U2, 128, 128, G, 0);
  k_gx<<<NB*128/256, 256, 0, stream>>>(G, Hb0, 128, 128, U1, NB*128);
  conv(U1, 128, Wc1, nullptr, nullptr, U2, 128, nullptr, nullptr, nullptr, NB, 128, 128, 0);
  conv(Hb0, 128, Wc1r, fS(2), fT(2), Yb, 128, U2, nullptr, nullptr, NB, 128, 128, 1);
  conv(Yb, 128, Wf1, nullptr, F(45), U1, 128, nullptr, nullptr, nullptr, NB, 128, 128, 0);
  k_se_gate<<<256, 256, 0, stream>>>(Yb, 128, 128, F(46), F(47), sgate);
  k_att_out<<<NB*128/256, 256, 0, stream>>>(U1, Yb, sgate, F(48),F(49),F(50),F(51), 128, Hb1, NB*128);

  // ---- gcn stage 2 (128 -> 256) ----
  conv(Hb1, 128, Wg2a, fS(3), fT(3), U1, 256, nullptr, nullptr, nullptr, NB, 128, 256, 0);
  conv(Hb1, 128, Wg2b, fS(4), fT(4), U2, 256, nullptr, nullptr, nullptr, NB, 128, 256, 0);
  k_gmat<<<5120, 256, 0, stream>>>(U1, U2, 256, 256, G, 1);
  k_gx<<<NB*128/256, 256, 0, stream>>>(G, Hb1, 128, 128, U1, NB*128);
  conv(U1, 128, Wc2, nullptr, nullptr, U2, 256, nullptr, nullptr, nullptr, NB, 128, 256, 0);
  conv(Hb1, 128, Wc2r, fS(5), fT(5), Yb, 256, U2, nullptr, nullptr, NB, 128, 256, 1);
  conv(Yb, 256, Wf2, nullptr, F(72), U1, 256, nullptr, nullptr, nullptr, NB, 256, 256, 0);
  k_se_gate<<<256, 256, 0, stream>>>(Yb, 256, 256, F(73), F(74), sgate);
  k_att_out<<<NB*256/256, 256, 0, stream>>>(U1, Yb, sgate, F(75),F(76),F(77),F(78), 256, Hb2, NB*256);

  // ---- gcn stage 3 (256 -> 256) ----
  conv(Hb2, 256, Wg3a, fS(6), fT(6), U1, 256, nullptr, nullptr, nullptr, NB, 256, 256, 0);
  conv(Hb2, 256, Wg3b, fS(7), fT(7), U2, 256, nullptr, nullptr, nullptr, NB, 256, 256, 0);
  k_gmat<<<5120, 256, 0, stream>>>(U1, U2, 256, 256, G, 1);
  k_gx<<<NB*256/256, 256, 0, stream>>>(G, Hb2, 256, 256, U1, NB*256);
  conv(U1, 256, Wc3, nullptr, nullptr, U2, 256, nullptr, nullptr, nullptr, NB, 256, 256, 0);
  conv(Hb2, 256, Wc3r, fS(8), fT(8), Yb, 256, U2, nullptr, nullptr, NB, 256, 256, 1);
  conv(Yb, 256, Wf3, nullptr, F(99), U1, 256, nullptr, nullptr, nullptr, NB, 256, 256, 0);
  k_se_gate<<<256, 256, 0, stream>>>(Yb, 256, 256, F(100), F(101), sgate);
  k_att_out<<<NB*256/256, 256, 0, stream>>>(U1, Yb, sgate, F(102),F(103),F(104),F(105), 256, Hb1, NB*256);

  // ---- residual from h1 + tem1, then tem_att ----
  conv(Hb0, 128, Wres, fS(9), fT(9), Hb2, 256, nullptr, Hb1, tem_tab, NB, 128, 256, 0);
  conv(Hb2, 256, Wta, fS(10), fT(10), Yb, 256, nullptr, nullptr, nullptr, NB, 256, 256, 0);

  // ---- local block ----
  k_pool<<<N2*256/256, 256, 0, stream>>>(Yb, Lt, N2*256);
  k_cnn1<<<N2*256/256, 256, 0, stream>>>(Lt, F(117), F(118), F(119),F(120),F(121),F(122), Z1, N2*256);
  conv(Z1, 256, Wcn2, fS(11), fT(11), Z2, 512, nullptr, nullptr, nullptr, N2, 256, 512, 1);
  conv(Lt, 256, Wlr, fS(12), fT(12), Lf, 512, nullptr, Z2, nullptr, N2, 256, 512, 0);

  // ---- global max pool + FC ----
  k_final<<<256, 512, 0, stream>>>(Lf, F(135), F(136), (float*)d_out);
}
